// Attention1_91139206021347
// MI455X (gfx1250) — compile-verified, matrix operand order checked by reading, not run on hardware
//
#include <hip/hip_runtime.h>
#include <hip/hip_bf16.h>
#include <stdint.h>

// ---------------------------------------------------------------------------
// Types for CDNA5 WMMA (wave32, 16x16x32 bf16 -> f32)
// ---------------------------------------------------------------------------
typedef __attribute__((ext_vector_type(16))) __bf16 v16bf;
typedef __attribute__((ext_vector_type(8)))  float  v8f;

// Problem constants
#define BATCH 8
#define SEQ 4
#define BS 32            // BATCH*SEQ
#define N1 196
#define N3 392           // concat tokens / keys
#define CDIM 384
#define HEADS 8
#define HD 48
#define HDP 64           // hd padded for clean K-chunks
#define NKP 448          // token/key dim padded to 7*64
#define MQ 12544         // BS*N3
#define MKV 6272         // BS*N1
#define SCALE 0.14433756729740643f  // 1/sqrt(48)

__device__ __forceinline__ unsigned short f2bf(float f) {
  union { float f; uint32_t u; } x; x.f = f;
  uint32_t u = x.u;
  u += 0x7FFFu + ((u >> 16) & 1u);   // round-to-nearest-even
  return (unsigned short)(u >> 16);
}

struct PairU4 { uint4 lo, hi; };
__device__ __forceinline__ v16bf make_frag(uint4 lo, uint4 hi) {
  PairU4 t{lo, hi};
  return __builtin_bit_cast(v16bf, t);
}

// Load one 16-element bf16 operand fragment for a 32-wide K chunk.
// ISA 16-bit A layout: lane-half h covers k = {kbase+8h..+7} then {kbase+16+8h..+7}
// -> two contiguous 16-byte loads. (B operand mirrored with N in lane%16.)
__device__ __forceinline__ v16bf load_frag(const unsigned short* row, int kbase, int half) {
  const unsigned short* p = row + kbase + 8 * half;
  uint4 lo = *(const uint4*)(p);
  uint4 hi = *(const uint4*)(p + 16);
  return make_frag(lo, hi);
}

__device__ __forceinline__ v8f wmma_bf16(v16bf a, v16bf b, v8f c) {
  // (neg_a, A, neg_b, B, c_mod, C, reuse_a, reuse_b)
  return __builtin_amdgcn_wmma_f32_16x16x32_bf16(false, a, false, b, (short)0, c, false, false);
}

// Async global -> LDS copy of one 16B chunk per lane (ASYNCcnt tracked).
// ldsOff is the byte offset inside the workgroup's dynamic LDS block (base 0,
// since there is no static LDS in these kernels).
__device__ __forceinline__ void async_g2l_b128(unsigned ldsOff, const void* gaddr) {
  unsigned long long g = (unsigned long long)gaddr;
  asm volatile("global_load_async_to_lds_b128 %0, %1, off"
               :: "v"(ldsOff), "v"(g) : "memory");
}
__device__ __forceinline__ void wait_asynccnt0() {
  asm volatile("s_wait_asynccnt 0x0" ::: "memory");
}

// ---------------------------------------------------------------------------
// fp32 -> bf16 conversion
// ---------------------------------------------------------------------------
__global__ void cvt_f32_bf16(const float* __restrict__ in,
                             unsigned short* __restrict__ out, int n) {
  int i = blockIdx.x * blockDim.x + threadIdx.x;
  if (i < n) out[i] = f2bf(in[i]);
}

// ---------------------------------------------------------------------------
// Generic WMMA GEMM:  C[M][N] = A[M][384] @ W[N][384]^T   (W row-major = B operand)
// Block: 128 threads = 4 waves; each wave one 16x16 tile; block covers 64(m) x 16(n).
// The 16x384 B tile is shared by all 4 waves -> staged once via async DMA to LDS.
// Epilogue scatters per MODE.
// ---------------------------------------------------------------------------
enum { MODE_KV = 0, MODE_Q = 1, MODE_PROJ = 2 };

#define GEMM_BTILE_BYTES (16 * CDIM * 2)   // 12288

template <int MODE>
__global__ void gemm_wmma(const unsigned short* __restrict__ A0,
                          const unsigned short* __restrict__ A1,
                          const unsigned short* __restrict__ Bw,
                          unsigned short* __restrict__ D0,
                          unsigned short* __restrict__ D1,
                          const float* __restrict__ bias,
                          float* __restrict__ Fout,
                          int keybase) {
  extern __shared__ char gsm[];
  unsigned short* Bs = (unsigned short*)gsm;   // [16][384] bf16

  // ---- async-stage the shared B tile (contiguous 12288 B) ----
  {
    const char* gsrc = (const char*)(Bw + (size_t)blockIdx.y * 16 * CDIM);
#pragma unroll
    for (int i = 0; i < GEMM_BTILE_BYTES / 2048; ++i) {   // 128 thr * 16B = 2048 B/step
      unsigned off = threadIdx.x * 16 + i * 2048;
      async_g2l_b128(off, gsrc + off);
    }
    wait_asynccnt0();
    __syncthreads();
  }

  const int lane = threadIdx.x & 31;
  const int wave = threadIdx.x >> 5;
  const int half = lane >> 4;
  const int l16  = lane & 15;
  const int m    = blockIdx.x * 64 + wave * 16 + l16;
  const int nb   = blockIdx.y * 16;

  const unsigned short* arow;
  if (MODE == MODE_Q) {
    // Q is taken over concat(x, y) along tokens
    int bs = m / N3, tok = m % N3;
    arow = (tok < N1) ? (A0 + (size_t)(bs * N1 + tok) * CDIM)
                      : (A1 + (size_t)(bs * N1 + tok - N1) * CDIM);
  } else {
    arow = A0 + (size_t)m * CDIM;
  }
  const unsigned short* brow = Bs + l16 * CDIM;   // LDS-resident B row

  v8f acc = {0.f, 0.f, 0.f, 0.f, 0.f, 0.f, 0.f, 0.f};
#pragma unroll 4
  for (int k = 0; k < CDIM; k += 32) {
    v16bf a = load_frag(arow, k, half);
    v16bf b = load_frag(brow, k, half);
    acc = wmma_bf16(a, b, acc);
  }

  const int mtop = blockIdx.x * 64 + wave * 16;
#pragma unroll
  for (int r = 0; r < 8; ++r) {
    int mm = mtop + r + 8 * half;       // C layout: M = r (+8 for upper lane half)
    int nn = nb + l16;                  // N = lane % 16
    float v = acc[r];
    if (MODE == MODE_KV) {
      int bs = mm / N1, tok = mm % N1;
      int two = nn / CDIM;              // nn in [0,768): 0 -> K, 1 -> V
      int rem = nn % CDIM;
      int h = rem / HD, d = rem % HD;
      if (two == 0)  // K: [bs][h][key][hd]
        D0[(((size_t)(bs * HEADS + h)) * NKP + keybase + tok) * HDP + d] = f2bf(v);
      else           // V^T: [bs][h][hd][key]
        D1[(((size_t)(bs * HEADS + h)) * HDP + d) * NKP + keybase + tok] = f2bf(v);
    } else if (MODE == MODE_Q) {
      int bs = mm / N3, tok = mm % N3;
      int h = nn / HD, d = nn % HD;
      D0[(((size_t)(bs * HEADS + h)) * NKP + tok) * HDP + d] = f2bf(v);
    } else {  // MODE_PROJ: fp32 + bias straight to output
      Fout[(size_t)mm * CDIM + nn] = v + bias[nn];
    }
  }
}

// ---------------------------------------------------------------------------
// Fused attention: per block = (bs, head, 64-row tile). 4 waves x 16 rows.
//   (entry)  async-DMA the whole per-head V^T block (64x448 bf16) into LDS,
//            overlapping phases 1-2; waited on only before phase 3.
//   phase 1: S = scale * Q K^T  (WMMA, fp32 scores -> LDS)
//   phase 2: softmax rows (mask padded keys), attn -> HBM (coalesced),
//            P (bf16) -> LDS
//   phase 3: ctx = P @ V  (WMMA against LDS-resident V^T), bf16 ctx -> ws
// Dynamic LDS layout (base 0):
//   S  [64][400] f32   @ 0        (102400 B)
//   P  [64][416] bf16  @ 102400   ( 53248 B)
//   inv[64]      f32   @ 155648   (   256 B)
//   VL [64][448] bf16  @ 155904   ( 57344 B)   total 213248 B
// ---------------------------------------------------------------------------
#define SMEM_S_ELEMS (64 * 400)
#define SMEM_P_ELEMS (64 * 416)
#define SMEM_P_OFF   (SMEM_S_ELEMS * 4)
#define SMEM_INV_OFF (SMEM_P_OFF + SMEM_P_ELEMS * 2)
#define SMEM_VL_OFF  (SMEM_INV_OFF + 64 * 4)
#define SMEM_VL_BYTES (64 * NKP * 2)
#define ATTN_SMEM_BYTES (SMEM_VL_OFF + SMEM_VL_BYTES)

__global__ void attn_kernel(const unsigned short* __restrict__ Qb,
                            const unsigned short* __restrict__ Kb,
                            const unsigned short* __restrict__ VTb,
                            float* __restrict__ attn_out,
                            unsigned short* __restrict__ Cb) {
  extern __shared__ char smem[];
  float* S = (float*)smem;                                     // [64][400]
  unsigned short* P = (unsigned short*)(smem + SMEM_P_OFF);    // [64][416]
  float* inv = (float*)(smem + SMEM_INV_OFF);                  // [64]
  unsigned short* VL = (unsigned short*)(smem + SMEM_VL_OFF);  // [64][448]

  const int blk = blockIdx.x;
  const int mt = blk % 7;
  const int h  = (blk / 7) % HEADS;
  const int bs = blk / (7 * HEADS);
  const int lane = threadIdx.x & 31;
  const int wave = threadIdx.x >> 5;
  const int half = lane >> 4;
  const int l16  = lane & 15;
  const int rowbase = mt * 64;

  const unsigned short* Qh = Qb  + ((size_t)(bs * HEADS + h)) * NKP * HDP;
  const unsigned short* Kh = Kb  + ((size_t)(bs * HEADS + h)) * NKP * HDP;
  const unsigned short* Vh = VTb + ((size_t)(bs * HEADS + h)) * HDP * NKP;

  // ---- kick off async V^T -> LDS copy; consumed in phase 3 ----
  {
    const char* gsrc = (const char*)Vh;
#pragma unroll
    for (int i = 0; i < SMEM_VL_BYTES / 2048; ++i) {   // 28 steps
      unsigned off = threadIdx.x * 16 + i * 2048;
      async_g2l_b128(SMEM_VL_OFF + off, gsrc + off);
    }
  }

  // ---- phase 1: scores ----
  const unsigned short* qrow = Qh + (size_t)(rowbase + wave * 16 + l16) * HDP;
  v16bf qa0 = load_frag(qrow, 0, half);
  v16bf qa1 = load_frag(qrow, 32, half);
  for (int kt = 0; kt < 25; ++kt) {     // 400 padded keys; >=392 masked later
    const unsigned short* krow = Kh + (size_t)(kt * 16 + l16) * HDP;
    v16bf kb0 = load_frag(krow, 0, half);
    v16bf kb1 = load_frag(krow, 32, half);
    v8f acc = {0.f, 0.f, 0.f, 0.f, 0.f, 0.f, 0.f, 0.f};
    acc = wmma_bf16(qa0, kb0, acc);
    acc = wmma_bf16(qa1, kb1, acc);
#pragma unroll
    for (int r = 0; r < 8; ++r)
      S[(wave * 16 + r + 8 * half) * 400 + kt * 16 + l16] = acc[r] * SCALE;
  }
  __syncthreads();

  // ---- phase 2a: row max / exp / sum ----
  if (threadIdx.x < 64) {
    float* Sr = S + threadIdx.x * 400;
    float mx = -1e30f;
    for (int c = 0; c < N3; ++c) mx = fmaxf(mx, Sr[c]);
    float sum = 0.f;
    for (int c = 0; c < N3; ++c) { float e = __expf(Sr[c] - mx); Sr[c] = e; sum += e; }
    inv[threadIdx.x] = 1.f / sum;
  }
  __syncthreads();

  // ---- phase 2b: normalize, write attn (coalesced) + bf16 P tile ----
  for (int idx = threadIdx.x; idx < SMEM_P_ELEMS; idx += 128) {
    int r = idx / 416, c = idx % 416;
    float p = (c < N3) ? S[r * 400 + c] * inv[r] : 0.f;
    P[idx] = f2bf(p);
    int grow = rowbase + r;
    if (c < N3 && grow < N3)
      attn_out[(((size_t)(bs * HEADS + h)) * N3 + grow) * N3 + c] = p;
  }
  // make the DMA'd V^T visible, then sync all waves
  wait_asynccnt0();
  __syncthreads();

  // ---- phase 3: ctx = P @ V (V^T rows now LDS-resident) ----
  const unsigned short* prow = P + (size_t)(wave * 16 + l16) * 416;
#pragma unroll
  for (int nt = 0; nt < 3; ++nt) {      // 48 hd cols = 3 n-tiles
    const unsigned short* vrow = VL + (size_t)(nt * 16 + l16) * NKP;
    v8f acc = {0.f, 0.f, 0.f, 0.f, 0.f, 0.f, 0.f, 0.f};
    for (int kt = 0; kt < 13; ++kt) {   // 416 padded keys / 32
      v16bf pa = load_frag(prow, kt * 32, half);
      v16bf vb = load_frag(vrow, kt * 32, half);
      acc = wmma_bf16(pa, vb, acc);
    }
#pragma unroll
    for (int r = 0; r < 8; ++r) {
      int grow = rowbase + wave * 16 + r + 8 * half;
      if (grow < N3) {
        int d = nt * 16 + l16;
        Cb[((size_t)bs * N3 + grow) * CDIM + h * HD + d] = f2bf(acc[r]);
      }
    }
  }
}

// ---------------------------------------------------------------------------
// Host launcher
// ---------------------------------------------------------------------------
extern "C" void kernel_launch(void* const* d_in, const int* in_sizes, int n_in,
                              void* d_out, int out_size, void* d_ws, size_t ws_size,
                              hipStream_t stream) {
  const float* x     = (const float*)d_in[0];   // [8,4,196,384]
  const float* y     = (const float*)d_in[1];   // [8,4,196,384]
  const float* Wqkv1 = (const float*)d_in[2];   // [768,384]
  const float* Wqkv2 = (const float*)d_in[3];   // [768,384]
  const float* Wqkv3 = (const float*)d_in[4];   // [384,384]
  const float* Wproj = (const float*)d_in[5];   // [384,384]
  const float* bproj = (const float*)d_in[6];   // [384]

  float* out0  = (float*)d_out;                      // [12544][384]
  float* attn0 = out0 + (size_t)MQ * CDIM;           // [32][8][392][392]

  char* ws = (char*)d_ws;
  size_t off = 0;
  auto take = [&](size_t bytes) -> char* {
    char* p = ws + off;
    off += (bytes + 255) & ~(size_t)255;
    return p;
  };
  const size_t qkBytes = (size_t)BS * HEADS * NKP * HDP * 2;
  unsigned short* Xb  = (unsigned short*)take((size_t)MKV * CDIM * 2);
  unsigned short* Yb  = (unsigned short*)take((size_t)MKV * CDIM * 2);
  unsigned short* W1b = (unsigned short*)take((size_t)768 * 384 * 2);
  unsigned short* W2b = (unsigned short*)take((size_t)768 * 384 * 2);
  unsigned short* W3b = (unsigned short*)take((size_t)384 * 384 * 2);
  unsigned short* Wpb = (unsigned short*)take((size_t)384 * 384 * 2);
  unsigned short* Qb  = (unsigned short*)take(qkBytes);
  unsigned short* Kb  = (unsigned short*)take(qkBytes);
  unsigned short* VTb = (unsigned short*)take(qkBytes);
  unsigned short* Cb  = (unsigned short*)take((size_t)MQ * CDIM * 2);

  // zero padded K/Q/V^T regions (graph-capture-safe)
  hipMemsetAsync(Qb, 0, qkBytes, stream);
  hipMemsetAsync(Kb, 0, qkBytes, stream);
  hipMemsetAsync(VTb, 0, qkBytes, stream);

  // fp32 -> bf16
  const int nxy = MKV * CDIM;
  cvt_f32_bf16<<<(nxy + 255) / 256, 256, 0, stream>>>(x, Xb, nxy);
  cvt_f32_bf16<<<(nxy + 255) / 256, 256, 0, stream>>>(y, Yb, nxy);
  cvt_f32_bf16<<<(768 * 384 + 255) / 256, 256, 0, stream>>>(Wqkv1, W1b, 768 * 384);
  cvt_f32_bf16<<<(768 * 384 + 255) / 256, 256, 0, stream>>>(Wqkv2, W2b, 768 * 384);
  cvt_f32_bf16<<<(384 * 384 + 255) / 256, 256, 0, stream>>>(Wqkv3, W3b, 384 * 384);
  cvt_f32_bf16<<<(384 * 384 + 255) / 256, 256, 0, stream>>>(Wproj, Wpb, 384 * 384);

  dim3 blk(128);
  // KV projections: M=6272, N=768
  gemm_wmma<MODE_KV><<<dim3(MKV / 64, 768 / 16), blk, GEMM_BTILE_BYTES, stream>>>(
      Xb, nullptr, W1b, Kb, VTb, nullptr, nullptr, 0);
  gemm_wmma<MODE_KV><<<dim3(MKV / 64, 768 / 16), blk, GEMM_BTILE_BYTES, stream>>>(
      Yb, nullptr, W2b, Kb, VTb, nullptr, nullptr, N1);
  // Q projection over concat(x,y): M=12544, N=384
  gemm_wmma<MODE_Q><<<dim3(MQ / 64, 384 / 16), blk, GEMM_BTILE_BYTES, stream>>>(
      Xb, Yb, W3b, Qb, nullptr, nullptr, nullptr, 0);

  // fused attention: 32 bs * 8 heads * 7 row tiles
  hipFuncSetAttribute(reinterpret_cast<const void*>(attn_kernel),
                      hipFuncAttributeMaxDynamicSharedMemorySize, ATTN_SMEM_BYTES);
  attn_kernel<<<BS * HEADS * 7, 128, ATTN_SMEM_BYTES, stream>>>(Qb, Kb, VTb, attn0, Cb);

  // output projection + bias: M=12544, N=384, fp32 out
  gemm_wmma<MODE_PROJ><<<dim3(MQ / 64, 384 / 16), blk, GEMM_BTILE_BYTES, stream>>>(
      Cb, nullptr, Wpb, nullptr, nullptr, bproj, out0, 0);
}
